// GRU_4483945857622
// MI455X (gfx1250) — matrix operand order repaired and checked by reading, not compiled, not run on hardware
//
#include <hip/hip_runtime.h>
#include <hip/hip_bf16.h>
#include <math.h>

// ---------------------------------------------------------------------------
// GRU classifier for MI455X (gfx1250, wave32, WMMA f32_16x16x32_f16).
//   B=256, S=512, F=128, H=64, 3H=192; MLP: 32768->512->256->100 + softmax
// Pipeline:
//   cvt(x,Wx,W1 -> f16)
//   k_gx        : gx = x @ Wx^T          (f16 in, f32 acc, f16 out)
//   k_gru       : 512-step scan, gx prefetched 1 step ahead, fast sigmoid/tanh
//   k_fc_split  : h1acc = outH @ W1^T, K split 16x into partial buffers
//   k_reduce    : h1 = relu(sum partials) -> f16
//   k_fc_h16    : h2 = relu(h1 @ W2^T)
//   k_fc_f32    : logits = h2 @ W3^T (N padded 112, guard 100)
//   k_softmax   : wave-per-row softmax
// ---------------------------------------------------------------------------

typedef __attribute__((ext_vector_type(16))) _Float16 v16h;
typedef __attribute__((ext_vector_type(8)))  _Float16 v8h;
typedef __attribute__((ext_vector_type(8)))  float    v8f;
typedef __attribute__((ext_vector_type(4)))  float    v4f;

#define WMMA16(a, b, c) \
  __builtin_amdgcn_wmma_f32_16x16x32_f16(false, (a), false, (b), (short)0, (c), false, false)

// --- A fragment (16x32 f16) from f16 row-major storage (global or LDS)
// lane l: m = m0+(l&15); elem i: k = k0+(l>>4)*16+i
__device__ __forceinline__ v16h frag_a_f16(const _Float16* __restrict__ A, int lda,
                                           int m0, int k0, int lane) {
  const int m = m0 + (lane & 15);
  const int half = lane >> 4;
  const _Float16* p = A + (size_t)m * lda + k0 + half * 16;
  v8h lo = *(const v8h*)(p);
  v8h hi = *(const v8h*)(p + 8);
  v16h r;
#pragma unroll
  for (int i = 0; i < 8; ++i) { r[i] = lo[i]; r[8 + i] = hi[i]; }
  return r;
}

// --- B fragment (32x16 f16): B = W^T, W row-major (N x K) f16.
// lane l: n = n0+(l&15); elem i: k = k0+(l>>4)*16+i  (one contiguous 32B run)
__device__ __forceinline__ v16h frag_b_h16(const _Float16* __restrict__ W, int ldw,
                                           int n0, int k0, int lane) {
  const int n = n0 + (lane & 15);
  const int half = lane >> 4;
  return *(const v16h*)(W + (size_t)n * ldw + k0 + half * 16);
}

// --- B fragment from f32 weights (small weights: W_h2h, W2)
__device__ __forceinline__ v16h frag_b_w32(const float* __restrict__ W, int ldw,
                                           int n0, int k0, int lane) {
  const int n = n0 + (lane & 15);
  const int half = lane >> 4;
  const float* p = W + (size_t)n * ldw + k0 + half * 16;
  v16h r;
#pragma unroll
  for (int i = 0; i < 16; i += 4) {
    v4f t = *(const v4f*)(p + i);
#pragma unroll
    for (int j = 0; j < 4; ++j) r[i + j] = (_Float16)t[j];
  }
  return r;
}

// guarded variant for ragged N (W3 has only 100 rows)
__device__ __forceinline__ v16h frag_b_w32_guard(const float* __restrict__ W, int ldw,
                                                 int n0, int nmax, int k0, int lane) {
  const int n = n0 + (lane & 15);
  v16h r;
  if (n >= nmax) {
#pragma unroll
    for (int i = 0; i < 16; ++i) r[i] = (_Float16)0.0f;
    return r;
  }
  const int half = lane >> 4;
  const float* p = W + (size_t)n * ldw + k0 + half * 16;
#pragma unroll
  for (int i = 0; i < 16; i += 4) {
    v4f t = *(const v4f*)(p + i);
#pragma unroll
    for (int j = 0; j < 4; ++j) r[i + j] = (_Float16)t[j];
  }
  return r;
}

__device__ __forceinline__ float fast_sigmoid(float x) {
  return __builtin_amdgcn_rcpf(1.0f + __expf(-x));
}
__device__ __forceinline__ float fast_tanh(float x) {
  // 1 - 2/(1+e^{2x}); exact at +-inf, 0 at 0
  return 1.0f - 2.0f * __builtin_amdgcn_rcpf(1.0f + __expf(2.0f * x));
}

// ---------------------------------------------------------------------------
// f32 -> f16 bulk convert, 8 elements per thread (b128 in, b128 out)
// ---------------------------------------------------------------------------
__global__ void __launch_bounds__(256) k_cvt(const float* __restrict__ src,
                                             _Float16* __restrict__ dst, int n8) {
  const int i = blockIdx.x * blockDim.x + threadIdx.x;
  if (i >= n8) return;
  const float* p = src + (size_t)i * 8;
  v4f a = *(const v4f*)(p);
  v4f b = *(const v4f*)(p + 4);
  v8h o;
#pragma unroll
  for (int j = 0; j < 4; ++j) { o[j] = (_Float16)a[j]; o[4 + j] = (_Float16)b[j]; }
  *(v8h*)(dst + (size_t)i * 8) = o;
}

// ---------------------------------------------------------------------------
// Kernel 1: gx = x @ Wx^T   (131072 x 128) @ (128 x 192) -> f16 (131072 x 192)
// pure f16 fragments, one 16x16 tile per wave, 4 K-steps
// ---------------------------------------------------------------------------
__global__ void __launch_bounds__(256) k_gx(const _Float16* __restrict__ xh,
                                            const _Float16* __restrict__ Wxh,
                                            _Float16* __restrict__ gx) {
  const int lane = threadIdx.x & 31;
  const int wave = blockIdx.x * (blockDim.x >> 5) + (threadIdx.x >> 5);
  const int NT = 12;
  if (wave >= 8192 * NT) return;
  const int mt = wave / NT, nt = wave % NT;
  const int m0 = mt * 16, n0 = nt * 16;
  v8f acc = {};
#pragma unroll
  for (int kk = 0; kk < 128; kk += 32) {
    v16h a = frag_a_f16(xh, 128, m0, kk, lane);
    v16h b = frag_b_h16(Wxh, 128, n0, kk, lane);
    acc = WMMA16(a, b, acc);
  }
  const int n = lane & 15, half = lane >> 4;
#pragma unroll
  for (int r = 0; r < 8; ++r)
    gx[(size_t)(m0 + r + 8 * half) * 192 + n0 + n] = (_Float16)acc[r];
}

// ---------------------------------------------------------------------------
// Kernel 2: sequential GRU scan. 16 blocks (16 batch rows each) x 4 waves.
// Wave w owns hidden cols [16w,16w+16): gh N-tiles {w, w+4, w+8}.
// gx is prefetched one timestep ahead to hide global latency under the WMMAs.
// ---------------------------------------------------------------------------
__global__ void __launch_bounds__(128) k_gru(const _Float16* __restrict__ gx,
                                             const float* __restrict__ Whh,
                                             _Float16* __restrict__ outH) {
  __shared__ _Float16 hsh[16 * 64];
  const int lane = threadIdx.x & 31;
  const int w = threadIdx.x >> 5;              // 0..3
  const int m0 = blockIdx.x * 16;              // batch-row base
  const int n = lane & 15, half = lane >> 4;

  for (int i = threadIdx.x; i < 16 * 64; i += 128) hsh[i] = (_Float16)0.0f;
  __syncthreads();

  // W_h2h fragments live in VGPRs for all 512 steps
  v16h bf[3][2];
#pragma unroll
  for (int p = 0; p < 3; ++p)
#pragma unroll
    for (int kh = 0; kh < 2; ++kh)
      bf[p][kh] = frag_b_w32(Whh, 64, (w + 4 * p) * 16, kh * 32, lane);

  // per-r row base pointers into gx
  const _Float16* gptr[8];
#pragma unroll
  for (int r = 0; r < 8; ++r)
    gptr[r] = gx + (size_t)(m0 + r + 8 * half) * 512 * 192;
  const int cr = (w + 0) * 16 + n;
  const int ci = (w + 4) * 16 + n;
  const int cn = (w + 8) * 16 + n;

  // preload gx for s = 0
  float cxr[8], cxi[8], cxn[8];
#pragma unroll
  for (int r = 0; r < 8; ++r) {
    const _Float16* g = gptr[r];
    cxr[r] = (float)g[cr]; cxi[r] = (float)g[ci]; cxn[r] = (float)g[cn];
  }

  v8f hreg = {};  // h[m0 + r + 8*half][16w + n]

  for (int s = 0; s < 512; ++s) {
    // prefetch gx for next step (clamped; overlaps WMMAs + gate math)
    const int sn = (s + 1 < 512) ? (s + 1) : 511;
    float nxr[8], nxi[8], nxn[8];
#pragma unroll
    for (int r = 0; r < 8; ++r) {
      const _Float16* g = gptr[r] + (size_t)sn * 192;
      nxr[r] = (float)g[cr]; nxi[r] = (float)g[ci]; nxn[r] = (float)g[cn];
    }

    v16h a0 = frag_a_f16(hsh, 64, 0, 0, lane);
    v16h a1 = frag_a_f16(hsh, 64, 0, 32, lane);
    v8f gr = {}, gi = {}, gn = {};
    gr = WMMA16(a0, bf[0][0], gr); gr = WMMA16(a1, bf[0][1], gr);
    gi = WMMA16(a0, bf[1][0], gi); gi = WMMA16(a1, bf[1][1], gi);
    gn = WMMA16(a0, bf[2][0], gn); gn = WMMA16(a1, bf[2][1], gn);

    v8f hnew;
#pragma unroll
    for (int r = 0; r < 8; ++r) {
      float rg = fast_sigmoid(cxr[r] + gr[r]);
      float ig = fast_sigmoid(cxi[r] + gi[r]);
      float ng = fast_tanh(cxn[r] + rg * gn[r]);
      hnew[r] = ng + ig * (hreg[r] - ng);
    }
    hreg = hnew;
#pragma unroll
    for (int r = 0; r < 8; ++r) { cxr[r] = nxr[r]; cxi[r] = nxi[r]; cxn[r] = nxn[r]; }

    __syncthreads();  // everyone consumed previous h
#pragma unroll
    for (int r = 0; r < 8; ++r) {
      const int mloc = r + 8 * half;
      const _Float16 hv = (_Float16)hnew[r];
      hsh[mloc * 64 + w * 16 + n] = hv;
      outH[(size_t)(m0 + mloc) * (512 * 64) + s * 64 + w * 16 + n] = hv;
    }
    __syncthreads();  // new h visible before next step's A loads
  }
}

// ---------------------------------------------------------------------------
// Kernel 3: K-split FC. Part[ks] += A(16-row tile) @ W^T(4 N-tiles), K/KSPLIT
// per wave. Partials are f32, summed by k_reduce. 2048 waves for layer 1.
// ---------------------------------------------------------------------------
template <int NTW, int KSPLIT>
__global__ void __launch_bounds__(256) k_fc_split(const _Float16* __restrict__ A,
                                                  const _Float16* __restrict__ Wh,
                                                  float* __restrict__ Part,
                                                  int M, int N, int K) {
  const int lane = threadIdx.x & 31;
  const int wave = blockIdx.x * (blockDim.x >> 5) + (threadIdx.x >> 5);
  const int mtiles = M >> 4;
  const int ngroups = (N >> 4) / NTW;
  if (wave >= mtiles * ngroups * KSPLIT) return;
  const int ks = wave / (mtiles * ngroups);
  const int rem = wave % (mtiles * ngroups);
  const int mt = rem / ngroups, ng = rem % ngroups;
  const int m0 = mt * 16;
  const int kchunk = K / KSPLIT;
  const int kbeg = ks * kchunk;

  v8f acc[NTW];
#pragma unroll
  for (int t = 0; t < NTW; ++t) acc[t] = (v8f){};

  for (int kk = kbeg; kk < kbeg + kchunk; kk += 32) {
    v16h a = frag_a_f16(A, K, m0, kk, lane);
#pragma unroll
    for (int t = 0; t < NTW; ++t) {
      v16h b = frag_b_h16(Wh, K, (ng * NTW + t) * 16, kk, lane);
      acc[t] = WMMA16(a, b, acc[t]);
    }
  }

  float* out = Part + (size_t)ks * M * N;
  const int n = lane & 15, half = lane >> 4;
#pragma unroll
  for (int t = 0; t < NTW; ++t) {
    const int n0 = (ng * NTW + t) * 16;
#pragma unroll
    for (int r = 0; r < 8; ++r)
      out[(size_t)(m0 + r + 8 * half) * N + n0 + n] = acc[t][r];
  }
}

// sum KSPLIT partials, relu, -> f16
template <int KSPLIT>
__global__ void __launch_bounds__(256) k_reduce(const float* __restrict__ Part,
                                                _Float16* __restrict__ Out, int MN) {
  const int i = blockIdx.x * blockDim.x + threadIdx.x;
  if (i >= MN) return;
  float s = 0.0f;
#pragma unroll
  for (int ks = 0; ks < KSPLIT; ++ks) s += Part[(size_t)ks * MN + i];
  Out[i] = (_Float16)(s > 0.0f ? s : 0.0f);
}

// ---------------------------------------------------------------------------
// Kernel 4: small FC, Out = relu(A @ W^T) f16, f32 weights converted inline
// ---------------------------------------------------------------------------
__global__ void __launch_bounds__(256) k_fc_h16(const _Float16* __restrict__ A,
                                                const float* __restrict__ W,
                                                _Float16* __restrict__ Out,
                                                int M, int N, int K) {
  const int lane = threadIdx.x & 31;
  const int wave = blockIdx.x * (blockDim.x >> 5) + (threadIdx.x >> 5);
  const int mtiles = M >> 4, ntiles = N >> 4;
  if (wave >= mtiles * ntiles) return;
  const int mt = wave / ntiles, nt = wave % ntiles;
  const int m0 = mt * 16, n0 = nt * 16;

  v8f acc = {};
  for (int kk = 0; kk < K; kk += 32) {
    v16h a = frag_a_f16(A, K, m0, kk, lane);
    v16h b = frag_b_w32(W, K, n0, kk, lane);
    acc = WMMA16(a, b, acc);
  }
  const int n = lane & 15, half = lane >> 4;
#pragma unroll
  for (int r = 0; r < 8; ++r) {
    float v = acc[r];
    v = v > 0.0f ? v : 0.0f;
    Out[(size_t)(m0 + r + 8 * half) * N + n0 + n] = (_Float16)v;
  }
}

// ---------------------------------------------------------------------------
// Kernel 5: final layer, logits = A @ W3^T, N padded to 112 (real 100), f32 out
// ---------------------------------------------------------------------------
__global__ void __launch_bounds__(256) k_fc_f32_guard(const _Float16* __restrict__ A,
                                                      const float* __restrict__ W,
                                                      float* __restrict__ Out,
                                                      int M, int Npad, int Nreal, int K) {
  const int lane = threadIdx.x & 31;
  const int wave = blockIdx.x * (blockDim.x >> 5) + (threadIdx.x >> 5);
  const int mtiles = M >> 4, ntiles = Npad >> 4;
  if (wave >= mtiles * ntiles) return;
  const int mt = wave / ntiles, nt = wave % ntiles;
  const int m0 = mt * 16, n0 = nt * 16;

  v8f acc = {};
  for (int kk = 0; kk < K; kk += 32) {
    v16h a = frag_a_f16(A, K, m0, kk, lane);
    v16h b = frag_b_w32_guard(W, K, n0, Nreal, kk, lane);
    acc = WMMA16(a, b, acc);
  }
  const int n = lane & 15, half = lane >> 4;
#pragma unroll
  for (int r = 0; r < 8; ++r)
    Out[(size_t)(m0 + r + 8 * half) * Npad + n0 + n] = acc[r];
}

// ---------------------------------------------------------------------------
// Kernel 6: row softmax over 100 valid columns (ld = 112), wave per row
// ---------------------------------------------------------------------------
__global__ void __launch_bounds__(256) k_softmax(const float* __restrict__ logits,
                                                 float* __restrict__ out, int rows) {
  const int lane = threadIdx.x & 31;
  const int row = blockIdx.x * (blockDim.x >> 5) + (threadIdx.x >> 5);
  if (row >= rows) return;
  const float* L = logits + (size_t)row * 112;

  float v[4];
  float mx = -INFINITY;
#pragma unroll
  for (int i = 0; i < 4; ++i) {
    const int c = lane + 32 * i;
    v[i] = (c < 100) ? L[c] : -INFINITY;
    mx = fmaxf(mx, v[i]);
  }
#pragma unroll
  for (int off = 16; off > 0; off >>= 1) mx = fmaxf(mx, __shfl_xor(mx, off, 32));

  float sum = 0.0f;
#pragma unroll
  for (int i = 0; i < 4; ++i) {
    const int c = lane + 32 * i;
    v[i] = (c < 100) ? __expf(v[i] - mx) : 0.0f;
    sum += v[i];
  }
#pragma unroll
  for (int off = 16; off > 0; off >>= 1) sum += __shfl_xor(sum, off, 32);

  const float inv = __builtin_amdgcn_rcpf(sum);
#pragma unroll
  for (int i = 0; i < 4; ++i) {
    const int c = lane + 32 * i;
    if (c < 100) out[(size_t)row * 100 + c] = v[i] * inv;
  }
}

// ---------------------------------------------------------------------------
extern "C" void kernel_launch(void* const* d_in, const int* in_sizes, int n_in,
                              void* d_out, int out_size, void* d_ws, size_t ws_size,
                              hipStream_t stream) {
  (void)in_sizes; (void)n_in; (void)out_size; (void)ws_size;
  const float* x   = (const float*)d_in[0];  // (256,512,128) = 16,777,216
  const float* Wx  = (const float*)d_in[1];  // (192,128)     = 24,576
  const float* Whh = (const float*)d_in[2];  // (192,64)
  const float* W1  = (const float*)d_in[3];  // (512,32768)   = 16,777,216
  const float* W2  = (const float*)d_in[4];  // (256,512)
  const float* W3  = (const float*)d_in[5];  // (100,256)
  float* out = (float*)d_out;                // (256,100)

  // workspace layout (~137 MB)
  char* ws = (char*)d_ws;
  _Float16* gx     = (_Float16*)(ws);                 // 50,331,648 B
  _Float16* outH   = (_Float16*)(ws + 50331648);      // 16,777,216 B
  _Float16* W1h    = (_Float16*)(ws + 67108864);      // 33,554,432 B
  _Float16* xh     = (_Float16*)(ws + 100663296);     // 33,554,432 B
  _Float16* Wxh    = (_Float16*)(ws + 134217728);     //     49,152 B
  float*    part   = (float*)   (ws + 134266880);     // 16*524,288 = 8,388,608 B
  _Float16* h1     = (_Float16*)(ws + 142655488);     //    262,144 B
  _Float16* h2     = (_Float16*)(ws + 142917632);     //    131,072 B
  float*    logits = (float*)   (ws + 143048704);     //    114,688 B

  // 0) one-shot f32 -> f16 conversions (8 elems / thread)
  k_cvt<<<8192, 256, 0, stream>>>(x,  xh,  16777216 / 8);
  k_cvt<<<12,   256, 0, stream>>>(Wx, Wxh, 24576 / 8);
  k_cvt<<<8192, 256, 0, stream>>>(W1, W1h, 16777216 / 8);

  // 1) gx = x @ Wx^T : 98304 wave-tiles
  k_gx<<<12288, 256, 0, stream>>>(xh, Wxh, gx);

  // 2) GRU scan: 16 batch tiles x 4 waves, 512 steps, gx prefetched
  k_gru<<<16, 128, 0, stream>>>(gx, Whh, outH);

  // 3) h1 = relu(outH @ W1^T): K=32768 split 16x -> 2048 waves, then reduce
  k_fc_split<4, 16><<<256, 256, 0, stream>>>(outH, W1h, part, 256, 512, 32768);
  k_reduce<16><<<512, 256, 0, stream>>>(part, h1, 256 * 512);

  // 4) h2 = relu(h1 @ W2^T): 256 waves, K=512
  k_fc_h16<<<32, 256, 0, stream>>>(h1, W2, h2, 256, 256, 512);

  // 5) logits = h2 @ W3^T : N padded 112 (real 100), K=256 -> 112 waves
  k_fc_f32_guard<<<14, 256, 0, stream>>>(h2, W3, logits, 256, 112, 100, 256);

  // 6) softmax over 100 cols, one wave per row
  k_softmax<<<32, 256, 0, stream>>>(logits, out, 256);
}
